// HPUAttentionImpl_18322330485425
// MI455X (gfx1250) — matrix operand order repaired and checked by reading, not compiled, not run on hardware
//
#include <hip/hip_runtime.h>

typedef __attribute__((ext_vector_type(16))) _Float16 v16h;
typedef __attribute__((ext_vector_type(8)))  _Float16 v8h;
typedef __attribute__((ext_vector_type(4)))  _Float16 v4h;
typedef __attribute__((ext_vector_type(2)))  _Float16 v2h;
typedef __attribute__((ext_vector_type(8)))  float    v8f;
typedef __attribute__((ext_vector_type(4)))  float    v4f;

namespace {
constexpr int S  = 2048;
constexpr int H  = 32;
constexpr int KV = 8;
constexpr int D  = 128;
constexpr int G  = H / KV;      // 4
constexpr int BM = 64;          // query rows per workgroup
constexpr int BN = 32;          // keys per iteration
}

__device__ inline unsigned pack2(float a, float b) {
  union { v2h h; unsigned u; } x;
  x.h[0] = (_Float16)a; x.h[1] = (_Float16)b;
  return x.u;
}

__global__ __launch_bounds__(128) void fa_alibi_kernel(
    const float* __restrict__ Q, const float* __restrict__ K,
    const float* __restrict__ V, const float* __restrict__ slopes,
    float* __restrict__ Out)
{
  __shared__ _Float16 Qs[BM * D];   // 16 KB, row-major [q][d], pre-scaled (log2e folded)
  __shared__ _Float16 Ks[BN * D];   //  8 KB, row-major [key][d]
  __shared__ _Float16 Vts[D * BN];  //  8 KB, transposed [d][key]

  const int tid  = threadIdx.x;
  const int wave = tid >> 5;
  const int lane = tid & 31;
  const int ln   = lane & 15;   // = query column within wave tile
  const int hh   = lane >> 4;   // half-wave index
  const int m0   = blockIdx.x * BM;
  const int h    = blockIdx.y;
  const int kvh  = h / G;
  // log2 domain: logits pre-multiplied by log2(e)
  const float qscale = 0.08838834764831845f * 1.44269504088896340736f;
  const float slope2 = slopes[h] * 1.44269504088896340736f;

  // ---- stage Q (f32 -> f16, pre-scaled), NT loads (read-once) ----
  for (int i = tid; i < BM * D / 4; i += 128) {
    const int row = i >> 5;
    const int c4  = (i & 31) * 4;
    const v4f f =
        __builtin_nontemporal_load((const v4f*)&Q[((m0 + row) * H + h) * D + c4]);
    v4h hv;
    hv[0] = (_Float16)(f[0] * qscale); hv[1] = (_Float16)(f[1] * qscale);
    hv[2] = (_Float16)(f[2] * qscale); hv[3] = (_Float16)(f[3] * qscale);
    *(v4h*)&Qs[row * D + c4] = hv;
  }
  __syncthreads();

  // ---- Q^T B-fragments in registers: B[k=d][n=q], lane n = ln ----
  v16h qb[4];
#pragma unroll
  for (int ks = 0; ks < 4; ++ks)
    qb[ks] = *(const v16h*)&Qs[(wave * 16 + ln) * D + ks * 32 + 16 * hh];

  // ---- accumulators / per-lane (per-query) softmax state ----
  v8f o[8];  // O^T: o[c][v] = O[d = c*16 + v + 8*hh][q = ln]
#pragma unroll
  for (int c = 0; c < 8; ++c)
#pragma unroll
    for (int e = 0; e < 8; ++e) o[c][e] = 0.0f;

  const int qg = m0 + wave * 16 + ln;          // this lane's query row
  float mi = -3.0e38f;                          // running max (scalar!)
  float li = 0.0f;                              // deferred per-lane partial sum
  float base = -slope2 * (float)qg;             // slope'*(n0 - qg), n0 = 0
  const float dbias16 = slope2 * 16.0f;
  const float dbias32 = slope2 * 32.0f;
  float kcv[8];
#pragma unroll
  for (int v = 0; v < 8; ++v) kcv[v] = slope2 * (float)(v + 8 * hh);

  const int nblocks = (m0 + BM) / BN;
  const int nfull   = m0 / BN;  // blocks fully below the diagonal (no mask)

  for (int nb = 0; nb < nblocks; ++nb) {
    const int n0 = nb * BN;
    __syncthreads();
    // ---- stage K (row-major) and V (transposed), f32 -> f16 ----
    for (int i = tid; i < BN * D / 4; i += 128) {
      const int row = i >> 5;
      const int c4  = (i & 31) * 4;
      const v4f fk = *(const v4f*)&K[((n0 + row) * KV + kvh) * D + c4];
      v4h hk;
      hk[0] = (_Float16)fk[0]; hk[1] = (_Float16)fk[1];
      hk[2] = (_Float16)fk[2]; hk[3] = (_Float16)fk[3];
      *(v4h*)&Ks[row * D + c4] = hk;
      const v4f fv = *(const v4f*)&V[((n0 + row) * KV + kvh) * D + c4];
      Vts[(c4 + 0) * BN + row] = (_Float16)fv[0];
      Vts[(c4 + 1) * BN + row] = (_Float16)fv[1];
      Vts[(c4 + 2) * BN + row] = (_Float16)fv[2];
      Vts[(c4 + 3) * BN + row] = (_Float16)fv[3];
    }
    __syncthreads();

    if (nb + 1 < nblocks) {  // hint next K/V tile into cache
      const int pr = tid & 31;
      __builtin_prefetch(&K[((n0 + BN + pr) * KV + kvh) * D + (tid >> 5) * 32], 0, 0);
      __builtin_prefetch(&V[((n0 + BN + pr) * KV + kvh) * D + (tid >> 5) * 32], 0, 0);
    }

    // ---- S^T = K Q^T : A = K tile rows (lane m = key), B = qb ----
    v16h kf[8];
#pragma unroll
    for (int ks = 0; ks < 4; ++ks) {
      const v8h a0 = *(const v8h*)&Ks[ln * D + ks * 32 + 16 * hh];
      const v8h a1 = *(const v8h*)&Ks[ln * D + ks * 32 + 16 * hh + 8];
      const v8h b0 = *(const v8h*)&Ks[(ln + 16) * D + ks * 32 + 16 * hh];
      const v8h b1 = *(const v8h*)&Ks[(ln + 16) * D + ks * 32 + 16 * hh + 8];
#pragma unroll
      for (int i = 0; i < 8; ++i) {
        kf[ks][i] = a0[i];     kf[ks][i + 8] = a1[i];
        kf[ks + 4][i] = b0[i]; kf[ks + 4][i + 8] = b1[i];
      }
    }
    v8f c0, c1;  // c0: keys n0+v+8hh ; c1: keys n0+16+v+8hh ; col = query ln
#pragma unroll
    for (int e = 0; e < 8; ++e) { c0[e] = 0.0f; c1[e] = 0.0f; }
#pragma unroll
    for (int ks = 0; ks < 4; ++ks) {
      c0 = __builtin_amdgcn_wmma_f32_16x16x32_f16(false, kf[ks], false, qb[ks],
                                                  (short)0, c0, false, false);
      c1 = __builtin_amdgcn_wmma_f32_16x16x32_f16(false, kf[ks + 4], false, qb[ks],
                                                  (short)0, c1, false, false);
    }

    // ---- ALiBi (incremental), causal mask on diagonal blocks only ----
    float s0[8], s1[8];
#pragma unroll
    for (int v = 0; v < 8; ++v) {
      const float bv = base + kcv[v];
      s0[v] = c0[v] + bv;
      s1[v] = c1[v] + (bv + dbias16);
    }
    base += dbias32;
    if (nb >= nfull) {  // wave-uniform branch; EXEC stays all-ones
      const int t = qg - n0 - 8 * hh;  // key > qg  <=>  v (+16) > t
#pragma unroll
      for (int v = 0; v < 8; ++v) {
        if (v > t)      s0[v] = -3.0e38f;
        if (v + 16 > t) s1[v] = -3.0e38f;
      }
    }

    // ---- per-query online softmax: in-lane tree + one cross-half shuffle ----
    float rm = fmaxf(s0[0], s1[0]);
#pragma unroll
    for (int v = 1; v < 8; ++v) rm = fmaxf(rm, fmaxf(s0[v], s1[v]));
    rm = fmaxf(rm, __shfl_xor(rm, 16));
    const float mc = fmaxf(mi, rm);
    const float al = __builtin_amdgcn_exp2f(mi - mc);
    mi = mc;

    float p0[8], p1[8];
    float rs = 0.0f;
#pragma unroll
    for (int v = 0; v < 8; ++v) {
      p0[v] = __builtin_amdgcn_exp2f(s0[v] - mc);
      p1[v] = __builtin_amdgcn_exp2f(s1[v] - mc);
      rs += p0[v] + p1[v];
    }
    li = li * al + rs;  // cross-lane sum deferred to epilogue

    // ---- build P^T B-fragment in registers (packed f16 + 4 half-swaps) ----
    union { unsigned u[8]; v16h hv; } pf;
    unsigned t0p[4], t1p[4];
#pragma unroll
    for (int j = 0; j < 4; ++j) {
      t0p[j] = pack2(p0[2 * j], p0[2 * j + 1]);
      t1p[j] = pack2(p1[2 * j], p1[2 * j + 1]);
    }
#pragma unroll
    for (int j = 0; j < 4; ++j) {
      const unsigned snd = hh ? t0p[j] : t1p[j];
      const unsigned rcv = (unsigned)__shfl_xor((int)snd, 16);
      pf.u[j]     = hh ? rcv : t0p[j];     // elems 0-7  = keys 16*hh + 0..7
      pf.u[4 + j] = hh ? t1p[j] : rcv;     // elems 8-15 = keys 16*hh + 8..15
    }

    // ---- rescale running O by scalar alpha ----
#pragma unroll
    for (int c = 0; c < 8; ++c)
#pragma unroll
      for (int e = 0; e < 8; ++e) o[c][e] *= al;

    // ---- O^T += V^T P^T : A = V^T rows (lane m = d), B = pf ----
    v16h va[8];
#pragma unroll
    for (int c = 0; c < 8; ++c) {
      const v8h a0 = *(const v8h*)&Vts[(c * 16 + ln) * BN + 16 * hh];
      const v8h a1 = *(const v8h*)&Vts[(c * 16 + ln) * BN + 16 * hh + 8];
#pragma unroll
      for (int i = 0; i < 8; ++i) { va[c][i] = a0[i]; va[c][i + 8] = a1[i]; }
    }
#pragma unroll
    for (int c = 0; c < 8; ++c)
      o[c] = __builtin_amdgcn_wmma_f32_16x16x32_f16(false, va[c], false, pf.hv,
                                                    (short)0, o[c], false, false);
  }

  // ---- epilogue: finish l reduction, normalize, vector NT stores ----
  li += __shfl_xor(li, 16);
  const float inv = __builtin_amdgcn_rcpf(li);
  float* orow = &Out[(qg * H + h) * D];
#pragma unroll
  for (int c = 0; c < 8; ++c) {
    v8f val;
#pragma unroll
    for (int e = 0; e < 8; ++e) val[e] = o[c][e] * inv;
    __builtin_nontemporal_store(val, (v8f*)&orow[c * 16 + 8 * hh]);
  }
}

extern "C" void kernel_launch(void* const* d_in, const int* in_sizes, int n_in,
                              void* d_out, int out_size, void* d_ws, size_t ws_size,
                              hipStream_t stream) {
  (void)in_sizes; (void)n_in; (void)out_size; (void)d_ws; (void)ws_size;
  const float* q  = (const float*)d_in[0];
  const float* k  = (const float*)d_in[1];
  const float* v  = (const float*)d_in[2];
  const float* sl = (const float*)d_in[3];
  float* out = (float*)d_out;
  dim3 grid(S / BM, H);
  dim3 block(128);
  fa_alibi_kernel<<<grid, block, 0, stream>>>(q, k, v, sl, out);
}
